// ERNNModel_25366076850554
// MI455X (gfx1250) — compile-verified, matrix operand order checked by reading, not run on hardware
//
#include <hip/hip_runtime.h>
#include <hip/hip_bf16.h>

// ---------------------------------------------------------------------------
// ERNN for MI455X (gfx1250, wave32, WMMA).
//  B=32, T=512, H=1024, E=256, V=128.
//  Pipeline:
//   1) weights -> bf16 ; r_const = Bm@default_h ; b_comb = W_hh@r_const + b_hh
//   2) W_comb = W_hh @ A (WMMA GEMM, bf16 out)   -> W_big = [A ; W_comb] (4096x1024)
//   3) X_emb  = concat(emb0[t0], emb1[t1]) bf16  ; gi_all = X_emb @ W_ih^T + b_ih
//   4) per t: G = h @ W_big^T + [r_const ; b_comb]  (one WMMA GEMM, TDM-staged h)
//             GRU elementwise -> h (bf16) and h_all[:, t]
//   5) dec0/dec1 = h_all @ decW^T + b  -> d_out
// ---------------------------------------------------------------------------

#define Bb 32
#define Tt 512
#define Hh 1024
#define Ee 256
#define Vv 128

typedef __attribute__((ext_vector_type(16))) __bf16 v16bf;
typedef __attribute__((ext_vector_type(8)))  __bf16 v8bf;
typedef __attribute__((ext_vector_type(8)))  float  v8f;
typedef __attribute__((ext_vector_type(4)))  unsigned int v4u;
typedef __attribute__((ext_vector_type(8)))  int    v8i;
typedef __attribute__((ext_vector_type(4)))  int    v4i;

#if defined(__has_builtin)
#if __has_builtin(__builtin_amdgcn_tensor_load_to_lds) && __has_builtin(__builtin_amdgcn_s_wait_tensorcnt)
#define HAVE_TDM 1
#endif
#endif

// ---------------- small helper kernels ----------------

__global__ void cvt_f32_bf16(const float* __restrict__ src, __bf16* __restrict__ dst, long long n) {
  long long i = (long long)blockIdx.x * blockDim.x + threadIdx.x;
  if (i < n) dst[i] = (__bf16)src[i];
}

__global__ void zero_bf16(__bf16* __restrict__ p, int n) {
  int i = blockIdx.x * blockDim.x + threadIdx.x;
  if (i < n) p[i] = (__bf16)0.0f;
}

// dst[r*n + c] = src[c*n + r]   (A^T in bf16)
__global__ void transpose_bf16(const float* __restrict__ src, __bf16* __restrict__ dst, int n) {
  int idx = blockIdx.x * blockDim.x + threadIdx.x;
  if (idx >= n * n) return;
  int r = idx / n, c = idx - r * n;
  dst[idx] = (__bf16)src[(size_t)c * n + r];
}

// out[i] = sum_k W[i*K+k]*x[k] (+bias)
__global__ void matvec_f32(const float* __restrict__ W, const float* __restrict__ x,
                           const float* __restrict__ bias, float* __restrict__ out,
                           int rows, int K) {
  int i = blockIdx.x * blockDim.x + threadIdx.x;
  if (i >= rows) return;
  float s = bias ? bias[i] : 0.0f;
  const float* w = W + (size_t)i * K;
  for (int k = 0; k < K; ++k) s += w[k] * x[k];
  out[i] = s;
}

// X_emb[bt, 0:512] = concat(emb0[tok0[bt]], emb1[tok1[bt]]) in bf16
__global__ void embed_concat(const int* __restrict__ t0, const int* __restrict__ t1,
                             const float* __restrict__ e0, const float* __restrict__ e1,
                             __bf16* __restrict__ X, int n) {
  int idx = blockIdx.x * blockDim.x + threadIdx.x;
  if (idx >= n) return;
  int bt = idx >> 9;          // / (2E = 512)
  int e  = idx & 511;
  float v = (e < Ee) ? e0[(size_t)t0[bt] * Ee + e]
                     : e1[(size_t)t1[bt] * Ee + (e - Ee)];
  X[idx] = (__bf16)v;
}

// ---------------- generic WMMA GEMM: C[M,N] = X[M,K] @ Y[N,K]^T (+bias) ----
// grid = (N/64, M/16), block = 128 (4 wave32, one 16x16 tile each).
// Fragment addressing follows the CDNA5 16-bit A/B VGPR layout:
//   lane l (0..15 -> row l, 16..31 -> row l-16 with K-offset +8),
//   half j in [0,16): K = (j&7) + 16*(j>>3) + 8*(lane>>4)  -> two 16B chunks.

__global__ void wmma_gemm(const __bf16* __restrict__ X, const __bf16* __restrict__ Y,
                          const float* __restrict__ bias,
                          float* __restrict__ outF, __bf16* __restrict__ outH,
                          int K, int ldx, int ldy, int ldc) {
  const int lane = threadIdx.x & 31;
  const int wv   = threadIdx.x >> 5;
  const int mt   = blockIdx.y * 16;
  const int nt   = blockIdx.x * 64 + wv * 16;
  const int hg   = lane >> 4;     // half-group: 0 or 1
  const int l15  = lane & 15;

  const __bf16* pa = X + (size_t)(mt + l15) * ldx + 8 * hg;
  const __bf16* pb = Y + (size_t)(nt + l15) * ldy + 8 * hg;

  v8f acc = {};
  union FragA { v16bf v; v8bf h[2]; };
  FragA fa, fb;
  for (int k0 = 0; k0 < K; k0 += 32) {
    fa.h[0] = *(const v8bf*)(pa + k0);
    fa.h[1] = *(const v8bf*)(pa + k0 + 16);
    fb.h[0] = *(const v8bf*)(pb + k0);
    fb.h[1] = *(const v8bf*)(pb + k0 + 16);
    acc = __builtin_amdgcn_wmma_f32_16x16x32_bf16(false, fa.v, false, fb.v,
                                                  (short)0, acc, false, false);
  }

  const int col   = nt + l15;
  const int rbase = mt + 8 * hg;      // C layout: VGPR r -> M = r + 8*(lane>>4)
  const float bv  = bias ? bias[col] : 0.0f;
#pragma unroll
  for (int r = 0; r < 8; ++r) {
    float v = acc[r] + bv;
    size_t o = (size_t)(rbase + r) * ldc + col;
    if (outF) outF[o] = v;
    else      outH[o] = (__bf16)v;
  }
}

// ---------------- per-step GEMM: G[32,4096] = h @ W_big^T + b_big ----------
// TDM stages the block's 16 rows of h (16 x 1024 bf16 = 32KB, contiguous) into
// LDS; A-fragments then come from ds_load_b128, B from global (cached in L2).

__global__ void step_gemm(const __bf16* __restrict__ h, const __bf16* __restrict__ Wbig,
                          const float* __restrict__ bias, float* __restrict__ G) {
  __shared__ __bf16 lds_h[16 * Hh];          // 32 KB of 320 KB/WGP
  const int lane = threadIdx.x & 31;
  const int wv   = threadIdx.x >> 5;
  const int mt   = blockIdx.y * 16;
  const int nt   = blockIdx.x * 64 + wv * 16;
  const int hg   = lane >> 4;
  const int l15  = lane & 15;

#if defined(HAVE_TDM)
  if (threadIdx.x < 32) {                    // one TDM issue per block (wave 0)
    unsigned long long ga = (unsigned long long)(const void*)(h + (size_t)mt * Hh);
    unsigned int lds_addr = (unsigned int)(unsigned long long)(const void*)&lds_h[0];
    // D# group0: count=1 | lds_addr | global_addr[56:0] | type=2 (bits 127:126)
    v4u g0 = { 1u, lds_addr,
               (unsigned int)(ga & 0xffffffffu),
               (unsigned int)((ga >> 32) & 0x01ffffffu) | (2u << 30) };
    // D# group1: 1 row of 4096 8-byte elements (data_size=3), tile==tensor
    v8i g1;
    g1[0] = 0x30000;                 // workgroup_mask=0, data_size=3 (8B)
    g1[1] = (int)(4096u << 16);      // tensor_dim0[15:0]  at bits [31:16]
    g1[2] = (int)(1u << 16);         // tensor_dim0 hi=0 ; tensor_dim1=1
    g1[3] = (int)(4096u << 16);      // tensor_dim1 hi=0 ; tile_dim0=4096
    g1[4] = 1;                       // tile_dim1=1 ; tile_dim2=0
    g1[5] = 4096;                    // tensor_dim0_stride (8B units)
    g1[6] = 0; g1[7] = 0;            // tensor_dim1_stride unused
    v4i gz4 = { 0, 0, 0, 0 };
    v8i gz8 = { 0, 0, 0, 0, 0, 0, 0, 0 };
    // 6-arg toolchain variant: (g0, g1, g2, g3, g4, cpol)
    __builtin_amdgcn_tensor_load_to_lds(g0, g1, gz4, gz4, gz8, 0);
    __builtin_amdgcn_s_wait_tensorcnt(0);
  }
#else
  {
    const unsigned long long* src = (const unsigned long long*)(h + (size_t)mt * Hh);
    unsigned long long* dst = (unsigned long long*)lds_h;
    for (int i = threadIdx.x; i < 16 * Hh / 4; i += 128) dst[i] = src[i];
  }
#endif
  __syncthreads();

  const __bf16* pa = lds_h + (size_t)l15 * Hh + 8 * hg;
  const __bf16* pb = Wbig + (size_t)(nt + l15) * Hh + 8 * hg;

  v8f acc = {};
  union FragS { v16bf v; v8bf h2[2]; };
  FragS fa, fb;
  for (int k0 = 0; k0 < Hh; k0 += 32) {
    __builtin_prefetch(pb + k0 + 256, 0, 0);      // global_prefetch next chunk
    fa.h2[0] = *(const v8bf*)(pa + k0);           // ds_load_b128 x2
    fa.h2[1] = *(const v8bf*)(pa + k0 + 16);
    fb.h2[0] = *(const v8bf*)(pb + k0);
    fb.h2[1] = *(const v8bf*)(pb + k0 + 16);
    acc = __builtin_amdgcn_wmma_f32_16x16x32_bf16(false, fa.v, false, fb.v,
                                                  (short)0, acc, false, false);
  }

  const int col   = nt + l15;
  const int rbase = mt + 8 * hg;
  const float bv  = bias[col];
#pragma unroll
  for (int r = 0; r < 8; ++r)
    G[(size_t)(rbase + r) * 4096 + col] = acc[r] + bv;
}

// ---------------- GRU elementwise update -----------------------------------
// G cols: [0,H)=new_h (+r_const), [H,2H)=gh_r, [2H,3H)=gh_z, [3H,4H)=gh_n
__global__ void gru_step(const float* __restrict__ G, const float* __restrict__ gi_all,
                         __bf16* __restrict__ h_cur, __bf16* __restrict__ h_all, int t) {
  int idx = blockIdx.x * blockDim.x + threadIdx.x;   // Bb*Hh threads
  int b = idx >> 10, j = idx & (Hh - 1);
  const float* g = G + (size_t)b * 4096;
  float nh = g[j];
  float gr = g[Hh + j], gz = g[2 * Hh + j], gn = g[3 * Hh + j];
  const float* gi = gi_all + ((size_t)b * Tt + t) * (3 * Hh);
  float r = 1.0f / (1.0f + __expf(-(gi[j] + gr)));
  float z = 1.0f / (1.0f + __expf(-(gi[Hh + j] + gz)));
  float n = tanhf(gi[2 * Hh + j] + r * gn);
  float hv = (1.0f - z) * n + z * nh;
  __bf16 hb = (__bf16)hv;
  h_cur[idx] = hb;
  h_all[((size_t)b * Tt + t) * Hh + j] = hb;
}

// ---------------------------------------------------------------------------

extern "C" void kernel_launch(void* const* d_in, const int* in_sizes, int n_in,
                              void* d_out, int out_size, void* d_ws, size_t ws_size,
                              hipStream_t stream) {
  (void)in_sizes; (void)n_in; (void)out_size; (void)ws_size;
  const int*   tok0 = (const int*)d_in[0];
  const int*   tok1 = (const int*)d_in[1];
  /* d_in[2] = conditions: unused by the reference */
  const float* emb0 = (const float*)d_in[3];
  const float* emb1 = (const float*)d_in[4];
  const float* W_ih = (const float*)d_in[5];
  const float* W_hh = (const float*)d_in[6];
  const float* b_ih = (const float*)d_in[7];
  const float* b_hh = (const float*)d_in[8];
  const float* A    = (const float*)d_in[9];
  const float* Bm   = (const float*)d_in[10];
  const float* defh = (const float*)d_in[11];
  const float* d0W  = (const float*)d_in[12];
  const float* d0b  = (const float*)d_in[13];
  const float* d1W  = (const float*)d_in[14];
  const float* d1b  = (const float*)d_in[15];
  float* out = (float*)d_out;

  // workspace carve-up
  char* w = (char*)d_ws;
  auto take = [&](size_t bytes) { char* p = w; w += (bytes + 255) & ~(size_t)255; return p; };
  float*  gi_all = (float*) take((size_t)Bb * Tt * 3 * Hh * 4);   // 201 MB
  __bf16* h_all  = (__bf16*)take((size_t)Bb * Tt * Hh * 2);       //  33 MB
  __bf16* W_big  = (__bf16*)take((size_t)4 * Hh * Hh * 2);        // [A ; W_comb]
  __bf16* Whh    = (__bf16*)take((size_t)3 * Hh * Hh * 2);
  __bf16* Whi    = (__bf16*)take((size_t)3 * Hh * 2 * Ee * 2);
  __bf16* A_T    = (__bf16*)take((size_t)Hh * Hh * 2);
  __bf16* Xemb   = (__bf16*)take((size_t)Bb * Tt * 2 * Ee * 2);
  __bf16* dW0    = (__bf16*)take((size_t)Vv * Hh * 2);
  __bf16* dW1    = (__bf16*)take((size_t)Vv * Hh * 2);
  float*  b_big  = (float*) take((size_t)4 * Hh * 4);             // [r_const ; b_comb]
  float*  G      = (float*) take((size_t)Bb * 4 * Hh * 4);
  __bf16* h_cur  = (__bf16*)take((size_t)Bb * Hh * 2);

  const int TPB = 256;
  auto blks = [](long long n, int t) { return (unsigned)((n + t - 1) / t); };

  // 1) weight conversions + bias folding
  cvt_f32_bf16<<<blks(3LL*Hh*2*Ee, TPB), TPB, 0, stream>>>(W_ih, Whi, 3LL*Hh*2*Ee);
  cvt_f32_bf16<<<blks(3LL*Hh*Hh,   TPB), TPB, 0, stream>>>(W_hh, Whh, 3LL*Hh*Hh);
  cvt_f32_bf16<<<blks((long long)Hh*Hh, TPB), TPB, 0, stream>>>(A, W_big, (long long)Hh*Hh); // rows 0..H-1 = A
  transpose_bf16<<<blks((long long)Hh*Hh, TPB), TPB, 0, stream>>>(A, A_T, Hh);
  cvt_f32_bf16<<<blks((long long)Vv*Hh, TPB), TPB, 0, stream>>>(d0W, dW0, (long long)Vv*Hh);
  cvt_f32_bf16<<<blks((long long)Vv*Hh, TPB), TPB, 0, stream>>>(d1W, dW1, (long long)Vv*Hh);
  matvec_f32<<<blks(Hh, TPB), TPB, 0, stream>>>(Bm, defh, nullptr, b_big, Hh, Hh);           // r_const
  matvec_f32<<<blks(3*Hh, TPB), TPB, 0, stream>>>(W_hh, b_big, b_hh, b_big + Hh, 3*Hh, Hh);  // b_comb

  // 2) W_comb = W_hh @ A  ->  W_big rows H..4H-1  (X @ Y^T with Y = A^T)
  wmma_gemm<<<dim3(Hh/64, 3*Hh/16), 128, 0, stream>>>(
      Whh, A_T, nullptr, nullptr, W_big + (size_t)Hh*Hh, Hh, Hh, Hh, Hh);

  // 3) embeddings + hoisted input projection: gi_all = X_emb @ W_ih^T + b_ih
  embed_concat<<<blks((long long)Bb*Tt*2*Ee, TPB), TPB, 0, stream>>>(
      tok0, tok1, emb0, emb1, Xemb, Bb*Tt*2*Ee);
  wmma_gemm<<<dim3(3*Hh/64, Bb*Tt/16), 128, 0, stream>>>(
      Xemb, Whi, b_ih, gi_all, nullptr, 2*Ee, 2*Ee, 2*Ee, 3*Hh);

  // 4) recurrence (serial critical path): one GEMM + one elementwise per step
  zero_bf16<<<blks(Bb*Hh, TPB), TPB, 0, stream>>>(h_cur, Bb*Hh);
  for (int t = 0; t < Tt; ++t) {
    step_gemm<<<dim3(4*Hh/64, Bb/16), 128, 0, stream>>>(h_cur, W_big, b_big, G);
    gru_step<<<blks(Bb*Hh, TPB), TPB, 0, stream>>>(G, gi_all, h_cur, h_all, t);
  }

  // 5) decoders
  wmma_gemm<<<dim3(Vv/64, Bb*Tt/16), 128, 0, stream>>>(
      h_all, dW0, d0b, out, nullptr, Hh, Hh, Hh, Vv);
  wmma_gemm<<<dim3(Vv/64, Bb*Tt/16), 128, 0, stream>>>(
      h_all, dW1, d1b, out + (size_t)Bb*Tt*Vv, nullptr, Hh, Hh, Hh, Vv);
}
